// GATEncoder_12738873000057
// MI455X (gfx1250) — matrix operand order repaired and checked by reading, not compiled, not run on hardware
//
#include <hip/hip_runtime.h>
#include <hip/hip_bf16.h>

// ---------------- problem constants (from reference) ----------------
#define NN 20000          // nodes
#define NE 640000         // edges
#define NT (NE + NN)      // edges + self loops = 660000
#define NH 4              // heads
#define NC 64             // channels/head
#define NF 256            // H*C
#define NL 3              // layers
#define NB 16             // batch segments
#define EPSN 1e-5f

// ---------------- CDNA5 async global->LDS path (guarded) ------------
#if __has_builtin(__builtin_amdgcn_global_load_async_to_lds_b128) && \
    __has_builtin(__builtin_amdgcn_s_wait_asynccnt)
#define GAT_ASYNC_LDS 1
#else
#define GAT_ASYNC_LDS 0
#endif

// pointer typedefs matching the builtin's V4i*1 / V4i*3 parameter types
typedef int v4i __attribute__((vector_size(16)));
typedef __attribute__((address_space(1))) v4i* as1_v4i;
typedef __attribute__((address_space(3))) v4i* as3_v4i;

// ---------------- vector typedefs for WMMA ----------------
typedef __attribute__((ext_vector_type(16))) __bf16 v16bf;
typedef __attribute__((ext_vector_type(8)))  __bf16 v8bf;
typedef __attribute__((ext_vector_type(8)))  float  v8f;

union BF16x16 { v16bf v; struct { v8bf lo, hi; } p; };

__device__ __forceinline__ __bf16 f2bf(float f) {
    unsigned u = __builtin_bit_cast(unsigned, f);
    unsigned r = (u + 0x7FFFu + ((u >> 16) & 1u)) >> 16;   // round-nearest-even
    return __builtin_bit_cast(__bf16, (unsigned short)r);
}

// order-preserving float<->uint encoding for atomic max over (possibly negative) floats
__device__ __forceinline__ unsigned enc_f32(float f) {
    unsigned b = __builtin_bit_cast(unsigned, f);
    return (b & 0x80000000u) ? ~b : (b | 0x80000000u);
}
__device__ __forceinline__ float dec_f32(unsigned k) {
    unsigned b = (k & 0x80000000u) ? (k & 0x7FFFFFFFu) : ~k;
    return __builtin_bit_cast(float, b);
}

// ---------------- weight prep: Wcat^T bf16 [512][256] ----------------
__global__ __launch_bounds__(256) void gat_prep_w(const float* __restrict__ Wl,
                                                  const float* __restrict__ Wr,
                                                  __bf16* __restrict__ wt) {
    int n = blockIdx.x;      // 0..511 (output column of [xl|xr])
    int k = threadIdx.x;     // 0..255 (input dim)
    float v = (n < NF) ? Wl[(size_t)k * NF + n] : Wr[(size_t)k * NF + (n - NF)];
    wt[(size_t)n * NF + k] = f2bf(v);
}

// ---------------- fused dual GEMM: out[N,512] = h @ [Wl|Wr] + [bl|br] ------
// block = 256 threads = 8 waves; block tile = 16 rows x 128 cols; each wave
// owns one 16x16 WMMA tile and walks K=256 in steps of 32.
// A-tile staged to LDS with GLOBAL_LOAD_ASYNC_TO_LDS_B128 (ASYNCcnt path),
// then converted fp32 -> bf16 ONCE into a second LDS tile so all 8 waves
// share the conversion work (16 converts/thread instead of 128).
__global__ __launch_bounds__(256) void gat_gemm_wmma(const float* __restrict__ A,
                                                     const __bf16* __restrict__ Wt,
                                                     const float* __restrict__ bl,
                                                     const float* __restrict__ br,
                                                     float* __restrict__ Out) {
    __shared__ __attribute__((aligned(16))) float  Af[16][264];  // fp32 stage (async dest)
    __shared__ __attribute__((aligned(32))) __bf16 Ab[16][272];  // bf16 tile for fragments
    const int m0  = blockIdx.y * 16;
    const int tid = threadIdx.x;
    const int row = tid >> 4;
    const int c0  = (tid & 15) * 16;

    // stage 16x256 fp32 A-tile into LDS
    {
        const float* src = A + (size_t)(m0 + row) * NF + c0;
#if GAT_ASYNC_LDS
        #pragma unroll
        for (int i = 0; i < 16; i += 4) {
            __builtin_amdgcn_global_load_async_to_lds_b128(
                (as1_v4i)(uintptr_t)(src + i),
                (as3_v4i)(unsigned)(uintptr_t)&Af[row][c0 + i],
                /*offset=*/0, /*cpol=*/0);
        }
        __builtin_amdgcn_s_wait_asynccnt(0);   // this wave's LDS writes landed
#else
        #pragma unroll
        for (int i = 0; i < 16; i += 4) {
            *(float4*)&Af[row][c0 + i] = *(const float4*)(src + i);
        }
#endif
    }
    __syncthreads();

    // one-shot fp32 -> bf16 conversion, shared by all waves
    {
        #pragma unroll
        for (int i = 0; i < 16; i += 4) {
            float4 f = *(const float4*)&Af[row][c0 + i];
            Ab[row][c0 + i + 0] = f2bf(f.x);
            Ab[row][c0 + i + 1] = f2bf(f.y);
            Ab[row][c0 + i + 2] = f2bf(f.z);
            Ab[row][c0 + i + 3] = f2bf(f.w);
        }
    }
    __syncthreads();

    const int wave    = tid >> 5;
    const int lane    = tid & 31;
    const int halfsel = (lane < 16) ? 0 : 1;         // wave32 fragment half
    const int arow    = lane & 15;                   // A row (M) for this lane
    const int ncol    = blockIdx.x * 128 + wave * 16 + (lane & 15);  // B/C column (N)

    v8f acc = {};
    #pragma unroll
    for (int kk = 0; kk < NF; kk += 32) {
        BF16x16 a;
        // A 16x32 bf16 layout: lanes 0-15 hold K [kk..kk+7] & [kk+16..kk+23],
        // lanes 16-31 hold K [kk+8..kk+15] & [kk+24..kk+31] of row M=lane%16.
        const int ak = kk + halfsel * 8;
        a.p.lo = *(const v8bf*)&Ab[arow][ak];
        a.p.hi = *(const v8bf*)&Ab[arow][ak + 16];
        // B 32x16 bf16 layout: same K order as A (lanes 0-15 hold K [kk..kk+7] & [kk+16..kk+23],
        // lanes 16-31 hold K [kk+8..kk+15] & [kk+24..kk+31]) of column N=lane%16.
        const __bf16* bp = Wt + (size_t)ncol * NF + kk + halfsel * 8;
        BF16x16 b;
        b.p.lo = *(const v8bf*)bp;
        b.p.hi = *(const v8bf*)(bp + 16);
        v16bf bv = b.v;
        if (kk + 32 < NF) __builtin_prefetch(bp + 32, 0, 1);  // global_prefetch_b8
        acc = __builtin_amdgcn_wmma_f32_16x16x32_bf16(
                  false, a.v, false, bv, (short)0, acc, false, false);
    }

    const float bias = (ncol < NF) ? bl[ncol] : br[ncol - NF];
    #pragma unroll
    for (int r = 0; r < 8; ++r) {  // C/D: VGPR r holds M = r + 8*halfsel
        int m = m0 + r + halfsel * 8;
        Out[(size_t)m * 512 + ncol] = acc[r] + bias;
    }
}

// ---------------- pass A: edge logits + segment max -----------------------
__global__ __launch_bounds__(256) void gat_edge_logits(const float* __restrict__ xlxr,
                                                       const int* __restrict__ ei,
                                                       const float* __restrict__ att,
                                                       float* __restrict__ ebuf,
                                                       unsigned* __restrict__ nmax) {
    const int wid = threadIdx.x >> 5, lane = threadIdx.x & 31;
    const int edge = blockIdx.x * 8 + wid;            // NT divisible by 8
    int s, d;
    if (edge < NE) { s = ei[edge]; d = ei[NE + edge]; } else { s = d = edge - NE; }
    const int c = lane * 8;
    const float* xl = xlxr + (size_t)s * 512 + c;
    const float* xr = xlxr + (size_t)d * 512 + NF + c;
    float4 a0 = *(const float4*)xl, a1 = *(const float4*)(xl + 4);
    float4 b0 = *(const float4*)xr, b1 = *(const float4*)(xr + 4);
    float av[8] = {a0.x, a0.y, a0.z, a0.w, a1.x, a1.y, a1.z, a1.w};
    float bv[8] = {b0.x, b0.y, b0.z, b0.w, b1.x, b1.y, b1.z, b1.w};
    float p = 0.f;
    #pragma unroll
    for (int i = 0; i < 8; ++i) {
        float t = av[i] + bv[i];
        t = (t > 0.f) ? t : 0.2f * t;                 // leaky_relu(0.2)
        p += att[c + i] * t;
    }
    p += __shfl_xor(p, 1, 8);
    p += __shfl_xor(p, 2, 8);
    p += __shfl_xor(p, 4, 8);
    if ((lane & 7) == 0) {
        int h = lane >> 3;
        ebuf[(size_t)edge * NH + h] = p;
        atomicMax(&nmax[(size_t)d * NH + h], enc_f32(p));
    }
}

// ---------------- pass B: exp + segment sum -------------------------------
__global__ __launch_bounds__(256) void gat_edge_exp(const int* __restrict__ ei,
                                                    float* __restrict__ ebuf,
                                                    const unsigned* __restrict__ nmax,
                                                    float* __restrict__ denom) {
    size_t i = (size_t)blockIdx.x * blockDim.x + threadIdx.x;
    if (i >= (size_t)NT * NH) return;
    int edge = (int)(i >> 2), h = (int)(i & 3);
    int d = (edge < NE) ? ei[NE + edge] : (edge - NE);
    float ex = __expf(ebuf[i] - dec_f32(nmax[(size_t)d * NH + h]));
    ebuf[i] = ex;
    atomicAdd(&denom[(size_t)d * NH + h], ex);
}

// ---------------- pass C: weighted scatter-add ----------------------------
__global__ __launch_bounds__(256) void gat_edge_accum(const float* __restrict__ xlxr,
                                                      const int* __restrict__ ei,
                                                      const float* __restrict__ ebuf,
                                                      const float* __restrict__ denom,
                                                      float* __restrict__ acc) {
    const int wid = threadIdx.x >> 5, lane = threadIdx.x & 31;
    const int edge = blockIdx.x * 8 + wid;
    int s, d;
    if (edge < NE) { s = ei[edge]; d = ei[NE + edge]; } else { s = d = edge - NE; }
    const int h = lane >> 3;
    const float alpha = ebuf[(size_t)edge * NH + h] / denom[(size_t)d * NH + h];
    const int c = lane * 8;
    const float* xl = xlxr + (size_t)s * 512 + c;
    float4 a0 = *(const float4*)xl, a1 = *(const float4*)(xl + 4);
    float av[8] = {a0.x, a0.y, a0.z, a0.w, a1.x, a1.y, a1.z, a1.w};
    float* dp = acc + (size_t)d * NF + c;
    #pragma unroll
    for (int i = 0; i < 8; ++i) atomicAdd(dp + i, alpha * av[i]);
}

// ---------------- GraphNorm statistics (per-feature over all N rows) ------
__global__ __launch_bounds__(256) void gat_norm_stats(const float* __restrict__ acc,
                                                      const float* __restrict__ bias,
                                                      float* __restrict__ sums,
                                                      float* __restrict__ sumsq) {
    const int c = threadIdx.x;
    const int r0 = blockIdx.x * 157;
    const int r1 = (r0 + 157 < NN) ? r0 + 157 : NN;
    float b = bias[c], s = 0.f, s2 = 0.f;
    for (int r = r0; r < r1; ++r) {
        float v = acc[(size_t)r * NF + c] + b;
        s += v; s2 += v * v;
    }
    atomicAdd(&sums[c], s);
    atomicAdd(&sumsq[c], s2);
}

__global__ void gat_norm_finalize(const float* __restrict__ sums,
                                  const float* __restrict__ sumsq,
                                  const float* __restrict__ w,
                                  const float* __restrict__ alpha,
                                  float* __restrict__ shift,
                                  float* __restrict__ scale) {
    const int c = threadIdx.x;
    float mean = sums[c] * (1.f / NN);
    float ex2  = sumsq[c] * (1.f / NN);
    float a = alpha[c];
    // E[(h - a*mean)^2] = E[h^2] - (2a - a^2) * mean^2
    float var = ex2 - (2.f * a - a * a) * mean * mean;
    shift[c] = a * mean;
    scale[c] = w[c] * rsqrtf(var + EPSN);
}

__global__ __launch_bounds__(256) void gat_norm_apply(const float* __restrict__ acc,
                                                      const float* __restrict__ bias,
                                                      const float* __restrict__ shift,
                                                      const float* __restrict__ scale,
                                                      const float* __restrict__ nb,
                                                      const float* __restrict__ prelu,
                                                      int l,
                                                      float* __restrict__ hout) {
    const int c = threadIdx.x;
    size_t i = (size_t)blockIdx.x * NF + c;
    float v = acc[i] + bias[c];
    float y = scale[c] * (v - shift[c]) + nb[c];
    float a = prelu[l];
    hout[i] = (y >= 0.f) ? y : a * y;
}

// ---------------- pooling, pooled-norm, fc --------------------------------
__global__ __launch_bounds__(256) void gat_pool(const float* __restrict__ h,
                                                const int* __restrict__ batch,
                                                float* __restrict__ pooled) {
    const int c = threadIdx.x, r = blockIdx.x;
    atomicAdd(&pooled[(size_t)batch[r] * NF + c], h[(size_t)r * NF + c]);
}

__global__ void gat_pool_norm(const float* __restrict__ pooled,
                              const float* __restrict__ w,
                              const float* __restrict__ nb,
                              const float* __restrict__ alpha,
                              float* __restrict__ pn) {
    const int c = threadIdx.x;
    float s = 0.f;
    #pragma unroll
    for (int r = 0; r < NB; ++r) s += pooled[r * NF + c];
    float mean = s * (1.f / NB);
    float sh = alpha[c] * mean;
    float s2 = 0.f;
    #pragma unroll
    for (int r = 0; r < NB; ++r) { float t = pooled[r * NF + c] - sh; s2 += t * t; }
    float sc = w[c] * rsqrtf(s2 * (1.f / NB) + EPSN);
    #pragma unroll
    for (int r = 0; r < NB; ++r) pn[r * NF + c] = sc * (pooled[r * NF + c] - sh) + nb[c];
}

__global__ __launch_bounds__(256) void gat_fc(const float* __restrict__ pn,
                                              const float* __restrict__ W,
                                              const float* __restrict__ b,
                                              float* __restrict__ out) {
    const int idx = blockIdx.x * 256 + threadIdx.x;   // 0..2047 = r*128 + j
    const int r = idx >> 7, j = idx & 127;
    float s = b[j];
    for (int c = 0; c < NF; ++c) s += pn[r * NF + c] * W[c * 128 + j];
    out[idx] = s;
}

// ---------------- host-side orchestration ---------------------------------
extern "C" void kernel_launch(void* const* d_in, const int* in_sizes, int n_in,
                              void* d_out, int out_size, void* d_ws, size_t ws_size,
                              hipStream_t stream) {
    const float* x     = (const float*)d_in[0];
    const int*   ei    = (const int*)d_in[1];
    const int*   batch = (const int*)d_in[2];
    const float* Wl    = (const float*)d_in[3];
    const float* bl    = (const float*)d_in[4];
    const float* Wr    = (const float*)d_in[5];
    const float* br    = (const float*)d_in[6];
    const float* att   = (const float*)d_in[7];
    const float* cbias = (const float*)d_in[8];
    const float* prelu = (const float*)d_in[9];
    const float* nw    = (const float*)d_in[10];
    const float* nbv   = (const float*)d_in[11];
    const float* nalph = (const float*)d_in[12];
    const float* fcW   = (const float*)d_in[13];
    const float* fcb   = (const float*)d_in[14];

    char* p = (char*)d_ws;
    auto take = [&p](size_t bytes) -> void* {
        void* r = (void*)p;
        p += (bytes + 255) & ~(size_t)255;
        return r;
    };
    float*    hbuf   = (float*)take((size_t)NN * NF * 4);
    float*    xlxr   = (float*)take((size_t)NN * 512 * 4);
    float*    acc    = (float*)take((size_t)NN * NF * 4);
    float*    ebuf   = (float*)take((size_t)NT * NH * 4);
    unsigned* nmax   = (unsigned*)take((size_t)NN * NH * 4);
    float*    denom  = (float*)take((size_t)NN * NH * 4);
    __bf16*   wt     = (__bf16*)take((size_t)512 * NF * 2);
    float*    sums   = (float*)take(NF * 4);
    float*    sumsq  = (float*)take(NF * 4);
    float*    shiftb = (float*)take(NF * 4);
    float*    scaleb = (float*)take(NF * 4);
    float*    pooled = (float*)take((size_t)NB * NF * 4);
    float*    pn     = (float*)take((size_t)NB * NF * 4);

    for (int l = 0; l < NL; ++l) {
        (void)hipMemsetAsync(acc,   0, (size_t)NN * NF * 4, stream);
        (void)hipMemsetAsync(nmax,  0, (size_t)NN * NH * 4, stream);  // 0 == encoded floor
        (void)hipMemsetAsync(denom, 0, (size_t)NN * NH * 4, stream);
        (void)hipMemsetAsync(sums,  0, NF * 4, stream);
        (void)hipMemsetAsync(sumsq, 0, NF * 4, stream);

        gat_prep_w<<<512, 256, 0, stream>>>(Wl + (size_t)l * NF * NF,
                                            Wr + (size_t)l * NF * NF, wt);
        const float* hin = (l == 0) ? x : hbuf;
        gat_gemm_wmma<<<dim3(4, NN / 16), 256, 0, stream>>>(hin, wt,
                                                            bl + l * NF, br + l * NF, xlxr);
        gat_edge_logits<<<NT / 8, 256, 0, stream>>>(xlxr, ei, att + l * NF, ebuf, nmax);
        gat_edge_exp<<<((size_t)NT * NH + 255) / 256, 256, 0, stream>>>(ei, ebuf, nmax, denom);
        gat_edge_accum<<<NT / 8, 256, 0, stream>>>(xlxr, ei, ebuf, denom, acc);
        gat_norm_stats<<<128, 256, 0, stream>>>(acc, cbias + l * NF, sums, sumsq);
        gat_norm_finalize<<<1, 256, 0, stream>>>(sums, sumsq, nw, nalph, shiftb, scaleb);
        gat_norm_apply<<<NN, 256, 0, stream>>>(acc, cbias + l * NF, shiftb, scaleb,
                                               nbv, prelu, l, hbuf);
    }

    (void)hipMemsetAsync(pooled, 0, (size_t)NB * NF * 4, stream);
    gat_pool<<<NN, 256, 0, stream>>>(hbuf, batch, pooled);
    gat_pool_norm<<<1, 256, 0, stream>>>(pooled, nw, nbv, nalph, pn);
    gat_fc<<<(NB * 128) / 256, 256, 0, stream>>>(pn, fcW, fcb, (float*)d_out);
}
